// SNNModel_66486093742253
// MI455X (gfx1250) — matrix operand order repaired and checked by reading, not compiled, not run on hardware
//
#include <hip/hip_runtime.h>

// SNN scan, CDNA5 (gfx1250, wave32). Weight-stationary WMMA formulation:
//   mem1^T[10x16b] += W1[10x6] * x_t^T[6x16b]        (one v_wmma_f32_16x16x32_f16)
//   mem2^T[27x16b] += W2[27x10] * spk1^T[10x16b]     (two v_wmma_f32_16x16x32_f16)
// Neurons on M, batch on N => WMMA D layout (batch striped across lanes) matches
// the next layer's B layout; neurons 8,9 are used in-lane by lanes 16-31 (no hop needed).
// x is streamed via double-buffered GLOBAL_LOAD_ASYNC_TO_LDS_B128 chunks
// (ASYNCcnt-pipelined), so the recurrence loop reads LDS, not HBM.

typedef __attribute__((ext_vector_type(16))) _Float16 v16h;
typedef __attribute__((ext_vector_type(8)))  float    v8f;

#define SNN_T      4096
#define SNN_BETA   0.9f
#define SNN_THRESH 1.0f

#define TC        16                  // timesteps per staged chunk
#define TCP       17                  // padded row pitch (words) -> bank-conflict-free
#define BUFW      (16 * 6 * TCP)      // words per buffer per wave (1632)
#define WAVES_PB  4                   // waves per block
#define LDS_BYTES (WAVES_PB * 2 * BUFW * 4)   // 52224 B dynamic LDS

extern __shared__ float lds_x[];

__device__ __forceinline__ float partner_xor16(float v) {
    int lane = (int)(threadIdx.x & 31u);
    int idx  = (lane ^ 16) << 2;                       // byte index for bpermute
    return __int_as_float(__builtin_amdgcn_ds_bpermute(idx, __float_as_int(v)));
}

// Issue one chunk copy: 12 wave-wide async b128 ops = 16b x 6c x 16t floats.
__device__ __forceinline__ void issue_chunk(const float* __restrict__ xrow,
                                            int t0, unsigned ldsWordBase) {
    const int lane = (int)(threadIdx.x & 31u);
    const int mr   = lane & 15;
    const int half = lane >> 4;
    #pragma unroll
    for (int i = 0; i < 12; ++i) {
        const int c = i >> 1;
        const int q = ((i & 1) << 1) | half;           // quad index 0..3
        const float* g = xrow + c * SNN_T + t0 + q * 4;
        unsigned lofs = (ldsWordBase + (unsigned)((mr * 6 + c) * TCP + q * 4)) * 4u;
        asm volatile("global_load_async_to_lds_b128 %0, %1, off"
                     :: "v"(lofs), "v"(g) : "memory");
    }
}

__global__ void __launch_bounds__(WAVES_PB * 32) snn_scan_wmma(
    const float* __restrict__ x,    // [B, 6, T]
    const float* __restrict__ W1,   // [10, 6]
    const float* __restrict__ b1,   // [10]
    const float* __restrict__ W2,   // [27, 10]
    const float* __restrict__ b2,   // [27]
    float* __restrict__ out,        // [B, 27]
    int B)
{
    const int lane = (int)(threadIdx.x & 31u);
    const int wave = (int)(threadIdx.x >> 5);
    const int tile = blockIdx.x * (int)(blockDim.x >> 5) + wave;
    const int b0   = tile * 16;
    if (b0 >= B) return;                               // uniform per wave
    const bool lo  = lane < 16;
    const int  mr  = lane & 15;

    const unsigned waveBase = (unsigned)(wave * 2 * BUFW);   // word offset of this wave's buffers

    // ---------------- constant A tiles (weights live in VGPRs) --------------
    // 16-bit A 16x32 layout: lanes 0-15 hold K=0..7 (h0..h7), lanes 16-31 K=8..15.
    v16h A1, A2a, A2b;
    #pragma unroll
    for (int i = 0; i < 16; ++i) {
        A1[i]  = (_Float16)0.f;
        A2a[i] = (_Float16)0.f;
        A2b[i] = (_Float16)0.f;
    }
    if (lo) {
        if (mr < 10) {
            #pragma unroll
            for (int c = 0; c < 6; ++c) A1[c] = (_Float16)W1[mr * 6 + c];
        }
        #pragma unroll
        for (int k = 0; k < 8; ++k)  A2a[k] = (_Float16)W2[mr * 10 + k];        // rows 0..15, K 0..7
        if (16 + mr < 27) {
            #pragma unroll
            for (int k = 0; k < 8; ++k) A2b[k] = (_Float16)W2[(16 + mr) * 10 + k]; // rows 16..26
        }
    } else {
        A2a[0] = (_Float16)W2[mr * 10 + 8];            // K=8,9 live in upper half-wave
        A2a[1] = (_Float16)W2[mr * 10 + 9];
        if (16 + mr < 27) {
            A2b[0] = (_Float16)W2[(16 + mr) * 10 + 8];
            A2b[1] = (_Float16)W2[(16 + mr) * 10 + 9];
        }
    }

    // ------------- bias tiles broadcast into C/D layout (M = neuron) --------
    v8f c1b, c2ba, c2bb;
    #pragma unroll
    for (int j = 0; j < 8; ++j) {
        int n1  = lo ? j : j + 8;                      // neuron row of this VGPR slot
        c1b[j]  = (n1 < 10) ? b1[n1] : 0.f;
        c2ba[j] = b2[n1];                              // rows 0..15 all valid
        int n2  = 16 + n1;
        c2bb[j] = (n2 < 27) ? b2[n2] : 0.f;
    }

    // ------------------------------ state -----------------------------------
    v8f m1, m2a, m2b;
    #pragma unroll
    for (int j = 0; j < 8; ++j) { m1[j] = 0.f; m2a[j] = 0.f; m2b[j] = 0.f; }

    v16h B1, B2;                                       // A is zero at K rows 16..31
    #pragma unroll
    for (int i = 0; i < 16; ++i) { B1[i] = (_Float16)0.f; B2[i] = (_Float16)0.f; }

    const float* xrow = x + (size_t)(b0 + mr) * (6 * SNN_T);

    // ---- prologue: stage chunk 0 into buffer 0 ------------------------------
    issue_chunk(xrow, 0, waveBase);

    const int NCHUNK = SNN_T / TC;                     // 256
    for (int ck = 0; ck < NCHUNK; ++ck) {
        if (ck + 1 < NCHUNK) {
            // overlap: start staging chunk ck+1, then fence chunk ck
            issue_chunk(xrow, (ck + 1) * TC, waveBase + (unsigned)(((ck + 1) & 1) * BUFW));
            if ((ck & 15) == 0)
                __builtin_prefetch(xrow + (ck + 64) * TC, 0, 1);  // warm L2 far ahead
            asm volatile("s_wait_asynccnt 0xc" ::: "memory");     // <=12 => chunk ck done
        } else {
            asm volatile("s_wait_asynccnt 0x0" ::: "memory");
        }
        const unsigned bufBase = waveBase + (unsigned)((ck & 1) * BUFW);

        #pragma unroll 2
        for (int dt = 0; dt < TC; ++dt) {
            // x_t^T into B1 from LDS: lane n (<16) holds channels 0..5 of batch b0+n
            #pragma unroll
            for (int c = 0; c < 6; ++c) {
                float v = lds_x[bufBase + (unsigned)((mr * 6 + c) * TCP + dt)];
                B1[c] = (_Float16)(lo ? v : 0.f);
            }

            // layer 1: mem1 = beta*mem1 - rst*thresh + b1 + W1 @ x_t^T
            v8f c1;
            #pragma unroll
            for (int j = 0; j < 8; ++j) {
                float r = (m1[j] > SNN_THRESH) ? SNN_THRESH : 0.f;
                c1[j] = SNN_BETA * m1[j] - r + c1b[j];
            }
            m1 = __builtin_amdgcn_wmma_f32_16x16x32_f16(
                     false, A1, false, B1, (short)0, c1, false, false);

            // layer-1 spikes into B2 rows: neurons 0..7 in lanes 0-15, 8..9 in lanes 16-31; partner copies meet zero A
            float s8 = partner_xor16(m1[0]);
            float s9 = partner_xor16(m1[1]);
            #pragma unroll
            for (int k = 0; k < 8; ++k)
                B2[k] = ((lo || k < 2) && (m1[k] > SNN_THRESH)) ? (_Float16)1.f : (_Float16)0.f;
            B2[8] = (lo && (s8 > SNN_THRESH)) ? (_Float16)1.f : (_Float16)0.f;
            B2[9] = (lo && (s9 > SNN_THRESH)) ? (_Float16)1.f : (_Float16)0.f;

            // layer 2 (two independent M tiles, same B)
            v8f c2a, c2b;
            #pragma unroll
            for (int j = 0; j < 8; ++j) {
                float ra = (m2a[j] > SNN_THRESH) ? SNN_THRESH : 0.f;
                float rb = (m2b[j] > SNN_THRESH) ? SNN_THRESH : 0.f;
                c2a[j] = SNN_BETA * m2a[j] - ra + c2ba[j];
                c2b[j] = SNN_BETA * m2b[j] - rb + c2bb[j];
            }
            m2a = __builtin_amdgcn_wmma_f32_16x16x32_f16(
                      false, A2a, false, B2, (short)0, c2a, false, false);
            m2b = __builtin_amdgcn_wmma_f32_16x16x32_f16(
                      false, A2b, false, B2, (short)0, c2b, false, false);
        }
    }

    // final-step spike of layer 2 -> out[B,27]
    float* op = out + (size_t)(b0 + mr) * 27;
    #pragma unroll
    for (int j = 0; j < 8; ++j) {
        int n = lo ? j : j + 8;                        // neurons 0..15 via tile a
        op[n] = (m2a[j] > SNN_THRESH) ? 1.f : 0.f;
        int n2 = 16 + n;                               // neurons 16..26 via tile b
        if (n2 < 27) op[n2] = (m2b[j] > SNN_THRESH) ? 1.f : 0.f;
    }
}

extern "C" void kernel_launch(void* const* d_in, const int* in_sizes, int n_in,
                              void* d_out, int out_size, void* d_ws, size_t ws_size,
                              hipStream_t stream) {
    const float* x  = (const float*)d_in[0];
    const float* W1 = (const float*)d_in[1];
    const float* b1 = (const float*)d_in[2];
    const float* W2 = (const float*)d_in[3];
    const float* b2 = (const float*)d_in[4];
    float* out = (float*)d_out;

    const int B = in_sizes[0] / (6 * SNN_T);           // 1024
    const int tiles = (B + 15) / 16;                   // 64 waves of work
    const int blocks = (tiles + WAVES_PB - 1) / WAVES_PB;

    snn_scan_wmma<<<blocks, WAVES_PB * 32, LDS_BYTES, stream>>>(x, W1, b1, W2, b2, out, B);
}
